// SpikingConv2D_70085276336948
// MI455X (gfx1250) — compile-verified, matrix operand order checked by reading, not run on hardware
//
#include <hip/hip_runtime.h>

typedef __attribute__((ext_vector_type(16))) _Float16 v16h;
typedef __attribute__((ext_vector_type(8)))  _Float16 v8h;
typedef __attribute__((ext_vector_type(4)))  _Float16 v4h;
typedef __attribute__((ext_vector_type(8)))  float    v8f;

#define BATCH 16
#define CIN   64
#define HH    112
#define WW    112
#define FOUT  128
#define YB    8          // output rows per workgroup (4 double-row iterations)
#define CPAD  72         // padded fastest dim: 144B rows -> 16B aligned, conflict-free b128
#define XPAD  114        // 112 + 2 halo columns

static __device__ inline v16h cat16(v8h lo, v8h hi) {
    return __builtin_shufflevector(lo, hi,
        0,1,2,3,4,5,6,7,8,9,10,11,12,13,14,15);
}

__global__ __launch_bounds__(256)
void spiking_conv3x3_wmma(const float* __restrict__ in,
                          const float* __restrict__ wgt,
                          const float* __restrict__ di,
                          float* __restrict__ out)
{
    // CDNA5 WGP: up to 320KB LDS per workgroup. Total here: 297,600 B.
    __shared__ _Float16 in_lds[4][XPAD][CPAD];   //  65 664 B : 4-row ring, ch-major f16
    __shared__ _Float16 w_lds[9][FOUT][CPAD];    // 165 888 B : all weights f16 [tap][f][c]
    __shared__ float    stage[8][16][17];        //   8 704 B : store-transpose stage
    __shared__ float    rawrow[2][CIN][WW];      //  57 344 B : async f32 row staging

    const int tid  = threadIdx.x;
    const int lane = tid & 31;
    const int wv   = tid >> 5;            // 8 waves = 2 row-groups x 4 N-groups
    const int rg   = wv >> 2;             // which of the 2 rows this wave computes
    const int ng   = wv & 3;              // which 32-channel N-slice (2 WMMA N-tiles)
    const int fbase = ng * 32;

    const int bb   = blockIdx.x / (HH / YB);
    const int y0   = (blockIdx.x % (HH / YB)) * YB;

    const int mrow = (lane & 15);         // fragment row for this lane
    const int koff = (lane & 16) ? 8 : 0; // K offset of upper half-wave (16-bit WMMA layout)

    // ---- stage all weights into LDS as f16: w_lds[tap][f][c] (K-contiguous B frags)
    for (int i = tid; i < FOUT * CIN; i += 256) {
        const int f = i >> 6;
        const int c = i & 63;
        const float* wp = wgt + (size_t)i * 9;
#pragma unroll
        for (int t = 0; t < 9; ++t)
            w_lds[t][f][c] = (_Float16)wp[t];
    }

    // ---- zero x-halo columns of all ring slots
    for (int i = tid; i < 4 * 2 * CIN; i += 256) {
        const int s = i / (2 * CIN);
        const int r = (i / CIN) & 1;
        const int c = i % CIN;
        in_lds[s][r ? (XPAD - 1) : 0][c] = (_Float16)0.0f;
    }

    // ---- async prefetch: raw f32 row -> LDS staging (overlaps with compute, ASYNCcnt)
    auto issue_row = [&](int r, int sl) {
        if ((unsigned)r >= HH) return;
        const uint32_t lbase = (uint32_t)(uintptr_t)&rawrow[sl][0][0];
        for (int q = tid; q < CIN * (WW / 4); q += 256) {
            const int c  = q / (WW / 4);
            const int x4 = (q % (WW / 4)) * 4;
            const uint64_t ga = (uint64_t)(uintptr_t)
                (in + (((size_t)bb * CIN + c) * HH + r) * WW + x4);
            const uint32_t la = lbase + (uint32_t)((c * WW + x4) * 4);
            asm volatile("global_load_async_to_lds_b128 %0, %1, off"
                         :: "v"(la), "v"(ga) : "memory");
        }
    };

    // ---- convert staged f32 row -> f16 channel-major ring slot (LDS -> LDS)
    auto convert_row = [&](int r, int sl) {
        if ((unsigned)r >= HH) return;
        const int slot = r & 3;
        for (int q = tid; q < (CIN / 4) * (WW / 4); q += 256) {
            const int cq = q / (WW / 4);
            const int x4 = (q % (WW / 4)) * 4;
            const int c0 = cq * 4;
            const float4 va = *(const float4*)&rawrow[sl][c0 + 0][x4];
            const float4 vb = *(const float4*)&rawrow[sl][c0 + 1][x4];
            const float4 vc = *(const float4*)&rawrow[sl][c0 + 2][x4];
            const float4 vd = *(const float4*)&rawrow[sl][c0 + 3][x4];
            const v4h p0 = {(_Float16)va.x, (_Float16)vb.x, (_Float16)vc.x, (_Float16)vd.x};
            const v4h p1 = {(_Float16)va.y, (_Float16)vb.y, (_Float16)vc.y, (_Float16)vd.y};
            const v4h p2 = {(_Float16)va.z, (_Float16)vb.z, (_Float16)vc.z, (_Float16)vd.z};
            const v4h p3 = {(_Float16)va.w, (_Float16)vb.w, (_Float16)vc.w, (_Float16)vd.w};
            *(v4h*)&in_lds[slot][x4 + 1][c0] = p0;
            *(v4h*)&in_lds[slot][x4 + 2][c0] = p1;
            *(v4h*)&in_lds[slot][x4 + 3][c0] = p2;
            *(v4h*)&in_lds[slot][x4 + 4][c0] = p3;
        }
    };

    // ---- prologue: land rows y0-1, y0, then start async for y0+1, y0+2
    issue_row(y0 - 1, 0);
    issue_row(y0, 1);
    asm volatile("s_wait_asynccnt 0x0" ::: "memory");
    __syncthreads();
    convert_row(y0 - 1, 0);
    convert_row(y0, 1);
    __syncthreads();                      // staging buffers free again
    issue_row(y0 + 1, 0);
    issue_row(y0 + 2, 1);

    const float thr0 = 1.0f - di[fbase + mrow];        // D_i[0][f], lower N-tile
    const float thr1 = 1.0f - di[fbase + 16 + mrow];   // upper N-tile

    const v8f vzero = {0.f,0.f,0.f,0.f,0.f,0.f,0.f,0.f};

#pragma unroll 1
    for (int yi = 0; yi < YB; yi += 2) {
        const int y = y0 + yi;
        asm volatile("s_wait_asynccnt 0x0" ::: "memory");  // this wave's asyncs arrived
        __syncthreads();                  // every wave waited + prior compute done (WAR)
        convert_row(y + 1, 0);
        convert_row(y + 2, 1);
        __syncthreads();                  // ring window {y-1..y+2} ready, staging free
        issue_row(y + 3, 0);              // prefetch next iteration under this compute
        issue_row(y + 4, 1);

        const int yr = y + rg;            // the row THIS wave computes

        v8f acc[7][2];
#pragma unroll
        for (int mt = 0; mt < 7; ++mt) { acc[mt][0] = vzero; acc[mt][1] = vzero; }

        // ---------------- 2D-register-tile implicit GEMM ----------------
        // per K-step: 2 B-frag + 7 A-frag loads -> 14 WMMAs
#pragma unroll
        for (int kh = 0; kh < 3; ++kh) {
            const int yy = yr + kh - 1;
            const unsigned yyu = (unsigned)__builtin_amdgcn_readfirstlane(yy);
            if (yyu >= HH) continue;      // scalar branch: EXEC stays all-ones
            const _Float16* rowp = &in_lds[yy & 3][0][0];
#pragma unroll
            for (int kw = 0; kw < 3; ++kw) {
                const _Float16* wtap = &w_lds[kh * 3 + kw][0][0];
#pragma unroll
                for (int cc = 0; cc < 2; ++cc) {
                    const int coff = cc * 32 + koff;
                    const _Float16* bp0 = wtap + (size_t)(fbase + mrow) * CPAD + coff;
                    const _Float16* bp1 = bp0 + (size_t)16 * CPAD;
                    const v16h b0 = cat16(*(const v8h*)bp0, *(const v8h*)(bp0 + 16));
                    const v16h b1 = cat16(*(const v8h*)bp1, *(const v8h*)(bp1 + 16));
#pragma unroll
                    for (int mt = 0; mt < 7; ++mt) {
                        const _Float16* p =
                            rowp + (size_t)(mt * 16 + mrow + kw) * CPAD + coff;
                        const v16h a = cat16(*(const v8h*)p, *(const v8h*)(p + 16));
                        acc[mt][0] = __builtin_amdgcn_wmma_f32_16x16x32_f16(
                            false, a, false, b0, (short)0, acc[mt][0], false, false);
                        acc[mt][1] = __builtin_amdgcn_wmma_f32_16x16x32_f16(
                            false, a, false, b1, (short)0, acc[mt][1], false, false);
                    }
                }
            }
        }

        // ---------------- fused epilogue: bias + clamp, transpose, coalesced stores ----
        // same-wave LDS ops execute in order (ISA 7.3): only compiler barriers needed
        const int crow = (lane & 16) ? 8 : 0;   // C/D layout: upper half-wave holds M+8
#pragma unroll
        for (int nt = 0; nt < 2; ++nt) {
            const float thr = nt ? thr1 : thr0;
#pragma unroll
            for (int mt = 0; mt < 7; ++mt) {
#pragma unroll
                for (int r = 0; r < 8; ++r) {
                    stage[wv][crow + r][mrow] = fminf(acc[mt][nt][r] + thr, 1.0f);
                }
                __asm__ volatile("" ::: "memory");   // keep store->load order
#pragma unroll
                for (int j = 0; j < 2; ++j) {
                    const int q  = lane + 32 * j;   // 64 float4 chunks of the 16x16 tile
                    const int fr = q >> 2;
                    const int c4 = (q & 3) * 4;
                    float4 v;
                    v.x = stage[wv][fr][c4 + 0];
                    v.y = stage[wv][fr][c4 + 1];
                    v.z = stage[wv][fr][c4 + 2];
                    v.w = stage[wv][fr][c4 + 3];
                    float* dst = out +
                        (((size_t)bb * FOUT + fbase + nt * 16 + fr) * HH + yr) * WW
                        + mt * 16 + c4;
                    *(float4*)dst = v;              // global_store_b128, x-contiguous
                }
                __asm__ volatile("" ::: "memory");   // reads retire before next overwrite
            }
        }
    }
}

extern "C" void kernel_launch(void* const* d_in, const int* in_sizes, int n_in,
                              void* d_out, int out_size, void* d_ws, size_t ws_size,
                              hipStream_t stream) {
    const float* tj  = (const float*)d_in[0];   // [16,64,112,112]
    const float* wgt = (const float*)d_in[1];   // [128,64,3,3]
    const float* di  = (const float*)d_in[2];   // [9,128], row 0 used
    float* out = (float*)d_out;                 // [16,128,112,112]

    dim3 grid(BATCH * (HH / YB));               // 224 workgroups
    dim3 block(256);                            // 8 wave32s
    hipLaunchKernelGGL(spiking_conv3x3_wmma, grid, block, 0, stream,
                       tj, wgt, di, out);
}
